// DBS_lstm_90220083020103
// MI455X (gfx1250) — matrix operand order repaired and checked by reading, not compiled, not run on hardware
//
#include <hip/hip_runtime.h>
#include <hip/hip_bf16.h>
#include <math.h>

typedef __bf16 v16bf __attribute__((ext_vector_type(16)));
typedef __bf16 v8bf  __attribute__((ext_vector_type(8)));
typedef __bf16 v4bf  __attribute__((ext_vector_type(4)));
typedef float  v8f   __attribute__((ext_vector_type(8)));
typedef int    v4i   __attribute__((ext_vector_type(4)));

#define T_FRAMES 512
#define HID      1024
#define GATES    4096
#define VOUT     20670

#define BM 64
#define BN 128
#define BK 32
#define LDT 40   // padded LDS pitch (bf16): 80B rows -> conflict-free b128 frag reads

// ---- gfx1250 async global->LDS copy support (probe-guarded) ----
#if defined(__gfx1250__) && __has_builtin(__builtin_amdgcn_global_load_async_to_lds_b128)
#define USE_ASYNC_LDS 1
#else
#define USE_ASYNC_LDS 0
#endif

#if __has_builtin(__builtin_amdgcn_s_wait_asynccnt)
#define WAIT_ASYNC(n) __builtin_amdgcn_s_wait_asynccnt(n)
#else
#define WAIT_ASYNC(n) asm volatile("s_wait_asynccnt %0" ::"n"(n))
#endif

#if USE_ASYNC_LDS
typedef __attribute__((address_space(1))) v4i* g4_t;   // global (prints as __device__)
typedef __attribute__((address_space(3))) v4i* l4_t;   // LDS
__device__ __forceinline__ void async_cp16(const __bf16* g, __bf16* l)
{
  // 16B per lane: global -> LDS, tracked by ASYNCcnt
  __builtin_amdgcn_global_load_async_to_lds_b128((g4_t)g, (l4_t)l, 0, 0);
}
#endif

// ---------------- shared WMMA tile compute ----------------
__device__ __forceinline__ void mma_tile(const __bf16* As, const __bf16* Ws,
                                         int wm, int wn, int half, int l16,
                                         v8f acc[2][2])
{
  // A frag: lane half 0 -> K {0..7,16..23}; half 1 -> K {8..15,24..31}
  const int kbA = half * 8;
  v16bf af[2], bfr[2];
#pragma unroll
  for (int i = 0; i < 2; ++i) {
    const __bf16* p = &As[(wm + i * 16 + l16) * LDT + kbA];
    v8bf lo = *(const v8bf*)p;
    v8bf hi = *(const v8bf*)(p + 16);
    af[i] = __builtin_shufflevector(lo, hi, 0,1,2,3,4,5,6,7,8,9,10,11,12,13,14,15);
  }
  // B frag: lane half 0 -> K {0..7,16..23}; half 1 -> K {8..15,24..31}
  const int kbB = half * 8;
#pragma unroll
  for (int j = 0; j < 2; ++j) {
    const __bf16* p = &Ws[(wn + j * 16 + l16) * LDT + kbB];
    v8bf lo = *(const v8bf*)p;
    v8bf hi = *(const v8bf*)(p + 16);
    bfr[j] = __builtin_shufflevector(lo, hi, 0,1,2,3,4,5,6,7,8,9,10,11,12,13,14,15);
  }
#pragma unroll
  for (int i = 0; i < 2; ++i)
#pragma unroll
    for (int j = 0; j < 2; ++j)
      acc[i][j] = __builtin_amdgcn_wmma_f32_16x16x32_bf16(
          false, af[i], false, bfr[j], (short)0, acc[i][j], false, false);
}

// ---------------- GEMM (bf16 inputs): C[M,N] = A[M,K]*W[N,K]^T + biases ----------------
// Full tiles: async global->LDS B128 copies, double-buffered & software-pipelined.
// Edge tiles: branchless clamped staging (rare: K=85 layer, one partial N tile of Wout).
__global__ __launch_bounds__(256) void gemm_bf16_nt(
    const __bf16* __restrict__ A, const __bf16* __restrict__ W,
    const float* __restrict__ bias0, const float* __restrict__ bias1,
    float* __restrict__ C, int M, int N, int K)
{
  __shared__ __align__(16) __bf16 As[2][BM * LDT];
  __shared__ __align__(16) __bf16 Ws[2][BN * LDT];

  const int tid  = threadIdx.x;
  const int lane = tid & 31;
  const int wave = tid >> 5;
  const int half = lane >> 4;
  const int l16  = lane & 15;

  const int m0 = blockIdx.y * BM;
  const int n0 = blockIdx.x * BN;
  const int wm = (wave >> 2) * 32;
  const int wn = (wave & 3) * 32;

  v8f acc[2][2];
#pragma unroll
  for (int i = 0; i < 2; ++i)
#pragma unroll
    for (int j = 0; j < 2; ++j)
#pragma unroll
      for (int r = 0; r < 8; ++r) acc[i][j][r] = 0.f;

  const int KT = (K + BK - 1) / BK;
  const bool fast = USE_ASYNC_LDS && (m0 + BM <= M) && (n0 + BN <= N) && ((K & 31) == 0);

  if (fast) {
#if USE_ASYNC_LDS
    // per-batch: each thread issues 1 A copy + 2 W copies (3 async instrs/wave)
    const int ra = tid >> 2, sa = tid & 3;              // A: 64 rows x 4 segs
    // prologue: stage tile 0 into buffer 0
    {
      async_cp16(&A[(size_t)(m0 + ra) * K + sa * 8], &As[0][ra * LDT + sa * 8]);
#pragma unroll
      for (int i = 0; i < 2; ++i) {
        int e = tid + i * 256, r = e >> 2, s = e & 3;   // W: 128 rows x 4 segs
        async_cp16(&W[(size_t)(n0 + r) * K + s * 8], &Ws[0][r * LDT + s * 8]);
      }
    }
    for (int kt = 0; kt < KT; ++kt) {
      const int cur = kt & 1;
      if (kt + 1 < KT) {
        const int k1 = (kt + 1) * BK;
        async_cp16(&A[(size_t)(m0 + ra) * K + k1 + sa * 8], &As[cur ^ 1][ra * LDT + sa * 8]);
#pragma unroll
        for (int i = 0; i < 2; ++i) {
          int e = tid + i * 256, r = e >> 2, s = e & 3;
          async_cp16(&W[(size_t)(n0 + r) * K + k1 + s * 8], &Ws[cur ^ 1][r * LDT + s * 8]);
        }
        WAIT_ASYNC(3);     // batch kt complete (next batch still in flight)
      } else {
        WAIT_ASYNC(0);
      }
      __syncthreads();
      mma_tile(As[cur], Ws[cur], wm, wn, half, l16, acc);
      __syncthreads();     // done reading buf[cur] before batch kt+2 overwrites it
    }
#endif
  } else {
    for (int kt = 0; kt < KT; ++kt) {
      const int k0 = kt * BK;
      // branchless clamped staging: all lanes load, select zero for OOB
      __bf16 va[(BM * BK) / 256];
#pragma unroll
      for (int i = 0; i < (BM * BK) / 256; ++i) {
        int e = tid + i * 256, r = e >> 5, k = e & 31;
        int gm = m0 + r, gk = k0 + k;
        bool ok = (gm < M) & (gk < K);
        int cm = ok ? gm : (M - 1);
        int ck = ok ? gk : 0;
        __bf16 v = A[(size_t)cm * K + ck];
        va[i] = ok ? v : (__bf16)0.f;
      }
      __bf16 vw[(BN * BK) / 256];
#pragma unroll
      for (int i = 0; i < (BN * BK) / 256; ++i) {
        int e = tid + i * 256, r = e >> 5, k = e & 31;
        int gn = n0 + r, gk = k0 + k;
        bool ok = (gn < N) & (gk < K);
        int cn = ok ? gn : (N - 1);
        int ck = ok ? gk : 0;
        __bf16 v = W[(size_t)cn * K + ck];
        vw[i] = ok ? v : (__bf16)0.f;
      }
#pragma unroll
      for (int i = 0; i < (BM * BK) / 256; ++i) {
        int e = tid + i * 256, r = e >> 5, k = e & 31;
        As[0][r * LDT + k] = va[i];
      }
#pragma unroll
      for (int i = 0; i < (BN * BK) / 256; ++i) {
        int e = tid + i * 256, r = e >> 5, k = e & 31;
        Ws[0][r * LDT + k] = vw[i];
      }
      __syncthreads();
      mma_tile(As[0], Ws[0], wm, wn, half, l16, acc);
      __syncthreads();
    }
  }

  // epilogue: lane<16 -> N=l16, M=r ; lane>=16 -> N=l16, M=r+8
#pragma unroll
  for (int i = 0; i < 2; ++i) {
#pragma unroll
    for (int j = 0; j < 2; ++j) {
      int n = n0 + wn + j * 16 + l16;
      if (n >= N) continue;
      float b = 0.f;
      if (bias0) b += bias0[n];
      if (bias1) b += bias1[n];
#pragma unroll
      for (int r = 0; r < 8; ++r) {
        int m = m0 + wm + i * 16 + (half ? r + 8 : r);
        if (m < M) C[(size_t)m * N + n] = acc[i][j][r] + b;
      }
    }
  }
}

// ---------------- fallback GEMM (fp32 inputs, convert while staging) ----------------
__global__ __launch_bounds__(256) void gemm_f32_nt(
    const float* __restrict__ A, const float* __restrict__ W,
    const float* __restrict__ bias0, const float* __restrict__ bias1,
    float* __restrict__ C, int M, int N, int K)
{
  __shared__ __align__(16) __bf16 As[BM * LDT];
  __shared__ __align__(16) __bf16 Ws[BN * LDT];

  const int tid  = threadIdx.x;
  const int lane = tid & 31;
  const int wave = tid >> 5;
  const int half = lane >> 4;
  const int l16  = lane & 15;
  const int m0 = blockIdx.y * BM;
  const int n0 = blockIdx.x * BN;
  const int wm = (wave >> 2) * 32;
  const int wn = (wave & 3) * 32;

  v8f acc[2][2];
#pragma unroll
  for (int i = 0; i < 2; ++i)
#pragma unroll
    for (int j = 0; j < 2; ++j)
#pragma unroll
      for (int r = 0; r < 8; ++r) acc[i][j][r] = 0.f;

  const int KT = (K + BK - 1) / BK;
  for (int kt = 0; kt < KT; ++kt) {
    const int k0 = kt * BK;
    float va[(BM * BK) / 256];
#pragma unroll
    for (int i = 0; i < (BM * BK) / 256; ++i) {
      int e = tid + i * 256, r = e >> 5, k = e & 31;
      int gm = m0 + r, gk = k0 + k;
      bool ok = (gm < M) & (gk < K);
      int cm = ok ? gm : (M - 1);
      int ck = ok ? gk : 0;
      float v = A[(size_t)cm * K + ck];
      va[i] = ok ? v : 0.f;
    }
    float vw[(BN * BK) / 256];
#pragma unroll
    for (int i = 0; i < (BN * BK) / 256; ++i) {
      int e = tid + i * 256, r = e >> 5, k = e & 31;
      int gn = n0 + r, gk = k0 + k;
      bool ok = (gn < N) & (gk < K);
      int cn = ok ? gn : (N - 1);
      int ck = ok ? gk : 0;
      float v = W[(size_t)cn * K + ck];
      vw[i] = ok ? v : 0.f;
    }
#pragma unroll
    for (int i = 0; i < (BM * BK) / 256; ++i) {
      int e = tid + i * 256, r = e >> 5, k = e & 31;
      As[r * LDT + k] = (__bf16)va[i];
    }
#pragma unroll
    for (int i = 0; i < (BN * BK) / 256; ++i) {
      int e = tid + i * 256, r = e >> 5, k = e & 31;
      Ws[r * LDT + k] = (__bf16)vw[i];
    }
    __syncthreads();
    mma_tile(As, Ws, wm, wn, half, l16, acc);
    __syncthreads();
  }

#pragma unroll
  for (int i = 0; i < 2; ++i) {
#pragma unroll
    for (int j = 0; j < 2; ++j) {
      int n = n0 + wn + j * 16 + l16;
      if (n >= N) continue;
      float b = 0.f;
      if (bias0) b += bias0[n];
      if (bias1) b += bias1[n];
#pragma unroll
      for (int r = 0; r < 8; ++r) {
        int m = m0 + wm + i * 16 + (half ? r + 8 : r);
        if (m < M) C[(size_t)m * N + n] = acc[i][j][r] + b;
      }
    }
  }
}

// ---------------- fp32 -> bf16 bulk conversion (n multiple of 4) ----------------
__global__ __launch_bounds__(256) void cvt_f32_bf16(const float* __restrict__ in,
                                                    __bf16* __restrict__ out, long n4)
{
  long i = (long)blockIdx.x * 256 + threadIdx.x;
  if (i < n4) {
    float4 v = ((const float4*)in)[i];
    v4bf o = { (__bf16)v.x, (__bf16)v.y, (__bf16)v.z, (__bf16)v.w };
    ((v4bf*)out)[i] = o;
  }
}

// ---------------- persistent grid-synced LSTM scan (exact fp32) ----------------
#define NB  64
#define UPB 16
#define RPB 64

__device__ __forceinline__ float sigmoidf(float x) { return 1.f / (1.f + __expf(-x)); }

__global__ __launch_bounds__(256) void lstm_scan(
    const float* __restrict__ Whh, const float* __restrict__ xg,
    float* __restrict__ hseq, float* __restrict__ hg,
    unsigned int* __restrict__ cnt)
{
  extern __shared__ float Wl[];      // RPB * HID fp32 = 256KB (<=320KB/WGP)
  __shared__ float hl[HID];
  __shared__ float rowsum[RPB];
  __shared__ float cstate[UPB];

  const int tid = threadIdx.x, lane = tid & 31, wave = tid >> 5, blk = blockIdx.x;

  for (int i = 0; i < (RPB * HID) / 256; ++i) {
    int e = tid + i * 256;
    int r = e >> 10, k = e & (HID - 1);
    int q = r >> 4, u = r & 15;
    Wl[r * HID + k] = Whh[(size_t)(q * HID + blk * UPB + u) * HID + k];
  }
  if (tid < UPB) cstate[tid] = 0.f;
  __syncthreads();

  for (int t = 0; t < T_FRAMES; ++t) {
    const int p = t & 1;
#pragma unroll
    for (int i = 0; i < HID / 256; ++i) {
      int e = tid + i * 256;
      hl[e] = (t == 0) ? 0.f : hg[p * HID + e];
    }
    __syncthreads();

#pragma unroll
    for (int rr = 0; rr < RPB / 8; ++rr) {
      int r = wave * 8 + rr;
      const float* wrow = &Wl[r * HID];
      float s = 0.f;
#pragma unroll
      for (int it = 0; it < HID / 64; ++it) {
        int k = it * 64 + lane * 2;
        s = fmaf(wrow[k],     hl[k],     s);
        s = fmaf(wrow[k + 1], hl[k + 1], s);
      }
#pragma unroll
      for (int off = 16; off > 0; off >>= 1) s += __shfl_xor(s, off, 32);
      if (lane == 0) rowsum[r] = s;
    }
    __syncthreads();

    if (tid < UPB) {
      int u = tid, col = blk * UPB + u;
      const float* xr = &xg[(size_t)t * GATES];
      float gi = sigmoidf(rowsum[ 0 + u] + xr[col]);
      float gf = sigmoidf(rowsum[16 + u] + xr[HID + col]);
      float gg = tanhf   (rowsum[32 + u] + xr[2 * HID + col]);
      float go = sigmoidf(rowsum[48 + u] + xr[3 * HID + col]);
      float c = gf * cstate[u] + gi * gg;
      cstate[u] = c;
      float h = go * tanhf(c);
      hseq[(size_t)t * HID + col] = h;
      hg[(p ^ 1) * HID + col]     = h;
    }
    __threadfence();
    __syncthreads();
    if (tid == 0) {
      __hip_atomic_fetch_add(cnt, 1u, __ATOMIC_RELEASE, __HIP_MEMORY_SCOPE_AGENT);
      const unsigned int target = (unsigned)NB * (unsigned)(t + 1);
      while (__hip_atomic_load(cnt, __ATOMIC_ACQUIRE, __HIP_MEMORY_SCOPE_AGENT) < target)
        __builtin_amdgcn_s_sleep(2);
    }
    __syncthreads();
  }
}

__global__ void zero_cnt(unsigned int* c) { if (threadIdx.x < 4) c[threadIdx.x] = 0u; }

// ---------------- host launch ----------------
extern "C" void kernel_launch(void* const* d_in, const int* in_sizes, int n_in,
                              void* d_out, int out_size, void* d_ws, size_t ws_size,
                              hipStream_t stream)
{
  const float* X    = (const float*)d_in[0];
  const float* Wih[3] = {(const float*)d_in[1], (const float*)d_in[5], (const float*)d_in[9]};
  const float* Whh[3] = {(const float*)d_in[2], (const float*)d_in[6], (const float*)d_in[10]};
  const float* bih[3] = {(const float*)d_in[3], (const float*)d_in[7], (const float*)d_in[11]};
  const float* bhh[3] = {(const float*)d_in[4], (const float*)d_in[8], (const float*)d_in[12]};
  const float* Wout = (const float*)d_in[13];
  float* out = (float*)d_out;

  // common workspace
  size_t off = 0;
  auto carve = [&](size_t bytes) { void* p = (char*)d_ws + off; off = (off + bytes + 255) & ~(size_t)255; return p; };
  float* xg = (float*)carve((size_t)T_FRAMES * GATES * 4);
  float* hA = (float*)carve((size_t)T_FRAMES * HID * 4);
  float* hB = (float*)carve((size_t)T_FRAMES * HID * 4);
  float* hg = (float*)carve(2 * HID * 4);
  unsigned int* cnt = (unsigned int*)carve(256);
  __bf16* hbf  = (__bf16*)carve((size_t)T_FRAMES * HID * 2);
  __bf16* xbf  = (__bf16*)carve((size_t)T_FRAMES * 85 * 2);
  __bf16* wbuf = (__bf16*)carve((size_t)VOUT * HID * 2);
  const bool bf16path = (off <= ws_size);

  const dim3 blk(256);
  const dim3 grid_gates((GATES + BN - 1) / BN, (T_FRAMES + BM - 1) / BM);
  const dim3 grid_out((VOUT + BN - 1) / BN, (T_FRAMES + BM - 1) / BM);
  const size_t scanLds = (size_t)RPB * HID * sizeof(float);
  auto cvt = [&](const float* src, __bf16* dst, long n) {
    long n4 = n >> 2;  // all sizes are multiples of 4
    cvt_f32_bf16<<<dim3((unsigned)((n4 + 255) / 256)), blk, 0, stream>>>(src, dst, n4);
  };

  zero_cnt<<<1, 32, 0, stream>>>(cnt);

  if (bf16path) {
    cvt(X, xbf, (long)T_FRAMES * 85);
    cvt(Wih[0], wbuf, (long)GATES * 85);
    gemm_bf16_nt<<<grid_gates, blk, 0, stream>>>(xbf, wbuf, bih[0], bhh[0], xg,
                                                 T_FRAMES, GATES, 85);
    lstm_scan<<<NB, blk, scanLds, stream>>>(Whh[0], xg, hA, hg, cnt + 0);

    cvt(hA, hbf, (long)T_FRAMES * HID);
    cvt(Wih[1], wbuf, (long)GATES * HID);
    gemm_bf16_nt<<<grid_gates, blk, 0, stream>>>(hbf, wbuf, bih[1], bhh[1], xg,
                                                 T_FRAMES, GATES, HID);
    lstm_scan<<<NB, blk, scanLds, stream>>>(Whh[1], xg, hB, hg, cnt + 1);

    cvt(hB, hbf, (long)T_FRAMES * HID);
    cvt(Wih[2], wbuf, (long)GATES * HID);
    gemm_bf16_nt<<<grid_gates, blk, 0, stream>>>(hbf, wbuf, bih[2], bhh[2], xg,
                                                 T_FRAMES, GATES, HID);
    lstm_scan<<<NB, blk, scanLds, stream>>>(Whh[2], xg, hA, hg, cnt + 2);

    cvt(hA, hbf, (long)T_FRAMES * HID);
    cvt(Wout, wbuf, (long)VOUT * HID);
    gemm_bf16_nt<<<grid_out, blk, 0, stream>>>(hbf, wbuf, nullptr, nullptr, out,
                                               T_FRAMES, VOUT, HID);
  } else {
    gemm_f32_nt<<<grid_gates, blk, 0, stream>>>(X, Wih[0], bih[0], bhh[0], xg,
                                                T_FRAMES, GATES, 85);
    lstm_scan<<<NB, blk, scanLds, stream>>>(Whh[0], xg, hA, hg, cnt + 0);
    gemm_f32_nt<<<grid_gates, blk, 0, stream>>>(hA, Wih[1], bih[1], bhh[1], xg,
                                                T_FRAMES, GATES, HID);
    lstm_scan<<<NB, blk, scanLds, stream>>>(Whh[1], xg, hB, hg, cnt + 1);
    gemm_f32_nt<<<grid_gates, blk, 0, stream>>>(hB, Wih[2], bih[2], bhh[2], xg,
                                                T_FRAMES, GATES, HID);
    lstm_scan<<<NB, blk, scanLds, stream>>>(Whh[2], xg, hA, hg, cnt + 2);
    gemm_f32_nt<<<grid_out, blk, 0, stream>>>(hA, Wout, nullptr, nullptr, out,
                                              T_FRAMES, VOUT, HID);
  }
}
